// EGNNDenoiser_70342974374465
// MI455X (gfx1250) — matrix operand order repaired and checked by reading, not compiled, not run on hardware
//
#include <hip/hip_runtime.h>
#include <hip/hip_bf16.h>

// ---------------------------------------------------------------------------
// EGNN layer for MI455X (gfx1250, wave32, WMMA).
// All GEMMs via v_wmma_f32_16x16x32_f16. Weights staged in LDS pre-swizzled
// into per-lane fragment-major order (one contiguous 32B LDS read per B
// fragment). Inter-GEMM transpose through f16 LDS scratch read back as two
// contiguous b128 chunks per lane. Coord head reduced in-register via
// shfl_xor butterflies.
// ---------------------------------------------------------------------------

typedef _Float16 v16h __attribute__((ext_vector_type(16)));
typedef _Float16 v8h  __attribute__((ext_vector_type(8)));
typedef float    v8f  __attribute__((ext_vector_type(8)));

#define DH 128   // hidden dim == feature dim

static __device__ __forceinline__ v8f wmma_f16(v16h a, v16h b, v8f c) {
  // 8 args: (neg_a, A, neg_b, B, c_mod, C, reuse_a, reuse_b)
  return __builtin_amdgcn_wmma_f32_16x16x32_f16(
      false, a, false, b, (short)0, c, false, false);
}

static __device__ __forceinline__ float silu_f(float v) {
  return v / (1.0f + __expf(-v));
}

// ---------------------------------------------------------------------------
// Fragment-major weight staging: WF[((kt*8+nt)*32 + lane)*16 + e] holds the
// element lane `lane` needs at position e of its B fragment for tile (kt,nt):
//   n = (lane&15) + nt*16 ,  k = kt*32 + 16*(e>=8) + (lane>>4)*8 + (e&7)
// K-rows >= Krows are zero-padded (matches zero-padded A k-tiles).
// ---------------------------------------------------------------------------
static __device__ __forceinline__ void stage_weights(_Float16* __restrict__ WF,
                                                     const float* __restrict__ W,
                                                     int KT, int Krows, int tid) {
  const int total = KT * 8 * 512;  // KT * NT(8) * 32 lanes * 16 elems
  for (int i = tid; i < total; i += 256) {
    int e  = i & 15;
    int ln = (i >> 4) & 31;
    int nt = (i >> 9) & 7;
    int kt = i >> 12;
    int n = (ln & 15) + nt * 16;
    int k = kt * 32 + ((e & 8) << 1) + (ln >> 4) * 8 + (e & 7);
    WF[i] = (k < Krows) ? (_Float16)W[k * DH + n] : (_Float16)0.0f;
  }
}

// One contiguous 32B LDS read per B fragment.
static __device__ __forceinline__ v16h load_b_frag(const _Float16* __restrict__ WF,
                                                   int kt, int nt, int lane) {
  return *(const v16h*)(WF + ((kt * 8 + nt) * 32 + lane) * 16);
}

// A fragment (16-bit 16x32 MxK) from one fp32 row in global memory.
// Element e of lane maps to K = cb + 16*(e>=8) + 8*(lane>=16) + (e&7).
static __device__ __forceinline__ v16h load_a_row(const float* __restrict__ row,
                                                  int cb, int half) {
  v16h a;
#pragma unroll
  for (int e = 0; e < 16; ++e) {
    int K = cb + ((e & 8) << 1) + half * 8 + (e & 7);
    a[e] = (_Float16)row[K];
  }
  return a;
}

// A fragment from f16 row-major [16][128] LDS scratch: two contiguous
// 16B chunks per lane.
static __device__ __forceinline__ v16h load_a_frag_sm(const _Float16* __restrict__ sm,
                                                      int kt, int lane) {
  int m = lane & 15, half = lane >> 4;
  const _Float16* p = sm + m * DH + kt * 32 + 8 * half;
  v8h lo = *(const v8h*)(p);
  v8h hi = *(const v8h*)(p + 16);
  return __builtin_shufflevector(lo, hi, 0, 1, 2, 3, 4, 5, 6, 7,
                                 8, 9, 10, 11, 12, 13, 14, 15);
}

// ---------------------------------------------------------------------------
// Zero workspace
// ---------------------------------------------------------------------------
__global__ void egnn_zero_kernel(float* __restrict__ p, long long n) {
  long long i = (long long)blockIdx.x * blockDim.x + threadIdx.x;
  long long stride = (long long)gridDim.x * blockDim.x;
  for (; i < n; i += stride) p[i] = 0.0f;
}

// ---------------------------------------------------------------------------
// Edge kernel: one wave = 16 edges (one M-tile). 8 waves / block.
// LDS: WF1 (9x8 frag tiles) + WF2 (4x8) + 8 * [16][128] f16 scratch = 139264 B
// ---------------------------------------------------------------------------
__global__ __launch_bounds__(256) void egnn_edge_kernel(
    const float* __restrict__ x, const float* __restrict__ pos,
    const int* __restrict__ ei, const float* __restrict__ eattr,
    const float* __restrict__ s,
    const float* __restrict__ We1, const float* __restrict__ be1,
    const float* __restrict__ We2, const float* __restrict__ be2,
    const float* __restrict__ Wc, const float* __restrict__ bc,
    float* __restrict__ msum, float* __restrict__ deg,
    float* __restrict__ cacc, int E) {
  extern __shared__ char smem[];
  _Float16* lwf1 = (_Float16*)smem;          // 9*8*512 = 36864 halves
  _Float16* lwf2 = lwf1 + 9 * 8 * 512;       // 4*8*512 = 16384 halves
  _Float16* scratch = lwf2 + 4 * 8 * 512;    // 8 waves * 2048 halves

  const int tid  = threadIdx.x;
  const int lane = tid & 31;
  const int wave = tid >> 5;
  const int half = lane >> 4;
  const int m    = lane & 15;

  stage_weights(lwf1, We1, 9, 258, tid);
  stage_weights(lwf2, We2, 4, 128, tid);
  __syncthreads();

  const int tile = blockIdx.x * 8 + wave;
  const int base = tile * 16;
  const bool active = base < E;
  _Float16* sm = scratch + wave * (16 * DH);

  v8f acc[8];

  if (active) {
    int eid = base + m;
    if (eid >= E) eid = E - 1;
    const int sidx = ei[eid];
    const int didx = ei[E + eid];
    const float dx = pos[didx * 3 + 0] - pos[sidx * 3 + 0];
    const float dy = pos[didx * 3 + 1] - pos[sidx * 3 + 1];
    const float dz = pos[didx * 3 + 2] - pos[sidx * 3 + 2];
    const float r2 = dx * dx + dy * dy + dz * dz;
    const float ea = eattr[eid];

    // C init = bias broadcast (C layout: N = 16*t + m, same for all 8 rows)
#pragma unroll
    for (int t = 0; t < 8; ++t) {
      float bias = be1[t * 16 + m];
      v8f c;
#pragma unroll
      for (int r = 0; r < 8; ++r) c[r] = bias;
      acc[t] = c;
    }

    const float* rowD = x + (size_t)didx * DH;
    const float* rowS = x + (size_t)sidx * DH;

    // GEMM1: k-tiles 0..3 = x[dst], 4..7 = x[src]
#pragma unroll
    for (int kt = 0; kt < 8; ++kt) {
      const float* row = (kt < 4) ? rowD : rowS;
      v16h a = load_a_row(row, (kt & 3) * 32, half);
#pragma unroll
      for (int nt = 0; nt < 8; ++nt)
        acc[nt] = wmma_f16(a, load_b_frag(lwf1, kt, nt, lane), acc[nt]);
    }
    // k-tile 8: [r2, edge_attr, 0...0]
    v16h a8;
#pragma unroll
    for (int e = 0; e < 16; ++e) {
      int K = ((e & 8) << 1) + half * 8 + (e & 7);
      float v = (K == 0) ? r2 : ((K == 1) ? ea : 0.0f);
      a8[e] = (_Float16)v;
    }
#pragma unroll
    for (int nt = 0; nt < 8; ++nt)
      acc[nt] = wmma_f16(a8, load_b_frag(lwf1, 8, nt, lane), acc[nt]);

    // SiLU, store h1 to per-wave scratch as f16 (C layout -> row-major)
#pragma unroll
    for (int t = 0; t < 8; ++t)
#pragma unroll
      for (int r = 0; r < 8; ++r)
        sm[(r + 8 * half) * DH + t * 16 + m] = (_Float16)silu_f(acc[t][r]);
  }
  __syncthreads();

  if (active) {
    // GEMM2: C init = bias
#pragma unroll
    for (int t = 0; t < 8; ++t) {
      float bias = be2[t * 16 + m];
      v8f c;
#pragma unroll
      for (int r = 0; r < 8; ++r) c[r] = bias;
      acc[t] = c;
    }
#pragma unroll
    for (int kt = 0; kt < 4; ++kt) {
      v16h a2 = load_a_frag_sm(sm, kt, lane);
#pragma unroll
      for (int nt = 0; nt < 8; ++nt)
        acc[nt] = wmma_f16(a2, load_b_frag(lwf2, kt, nt, lane), acc[nt]);
    }

    // Per-row scalars for the 8 edge rows this lane owns in C layout
    float sg[8];
    int dsts[8];
    bool ev[8];
#pragma unroll
    for (int r = 0; r < 8; ++r) {
      int e2 = base + 8 * half + r;
      ev[r] = e2 < E;
      if (e2 >= E) e2 = E - 1;
      sg[r]   = s[ei[e2]];
      dsts[r] = ei[E + e2];
    }

    // SiLU + s-gate (keep gated values in acc), atomic scatter into m_sum
#pragma unroll
    for (int t = 0; t < 8; ++t) {
#pragma unroll
      for (int r = 0; r < 8; ++r) {
        float v = silu_f(acc[t][r]) * sg[r];
        acc[t][r] = v;
        if (ev[r])
          atomicAdd(&msum[(size_t)dsts[r] * DH + t * 16 + m], v);
      }
    }

    // gamma = m . W_c + b_c, fully in-register:
    // lane partial over its 8 owned columns, butterfly over 16-lane group.
    float p[8];
#pragma unroll
    for (int r = 0; r < 8; ++r) p[r] = 0.0f;
#pragma unroll
    for (int t = 0; t < 8; ++t) {
      float wc = Wc[t * 16 + m];
#pragma unroll
      for (int r = 0; r < 8; ++r) p[r] += acc[t][r] * wc;
    }
#pragma unroll
    for (int mask = 1; mask < 16; mask <<= 1) {
#pragma unroll
      for (int r = 0; r < 8; ++r) p[r] += __shfl_xor(p[r], mask, 32);
    }
    // lanes 0..7 handle rows 0..7, lanes 16..23 handle rows 8..15
    if (m < 8) {
      int row = 8 * half + m;
      int eid = base + row;
      if (eid < E) {
        float g = p[0];
#pragma unroll
        for (int r = 1; r < 8; ++r) g = (m == r) ? p[r] : g;
        float gamma = g + bc[0];
        int ss = ei[eid];
        int dd = ei[E + eid];
        float dx = pos[dd * 3 + 0] - pos[ss * 3 + 0];
        float dy = pos[dd * 3 + 1] - pos[ss * 3 + 1];
        float dz = pos[dd * 3 + 2] - pos[ss * 3 + 2];
        atomicAdd(&cacc[dd * 3 + 0], gamma * dx);
        atomicAdd(&cacc[dd * 3 + 1], gamma * dy);
        atomicAdd(&cacc[dd * 3 + 2], gamma * dz);
        atomicAdd(&deg[dd], 1.0f);
      }
    }
  }
}

// ---------------------------------------------------------------------------
// Node kernel: one wave = 16 nodes. A = [x | m_sum/deg] (K=256).
// LDS: WF1 (8x8) + WF2 (4x8) + scratch = 131072 B
// ---------------------------------------------------------------------------
__global__ __launch_bounds__(256) void egnn_node_kernel(
    const float* __restrict__ x, const float* __restrict__ pos,
    const float* __restrict__ Wn1, const float* __restrict__ bn1,
    const float* __restrict__ Wn2, const float* __restrict__ bn2,
    const float* __restrict__ msum, const float* __restrict__ deg,
    const float* __restrict__ cacc,
    float* __restrict__ xout, float* __restrict__ posout, int N) {
  extern __shared__ char smem[];
  _Float16* lwf1 = (_Float16*)smem;          // 8*8*512 = 32768 halves
  _Float16* lwf2 = lwf1 + 8 * 8 * 512;       // 16384 halves
  _Float16* scratch = lwf2 + 4 * 8 * 512;    // 8 waves * 2048 halves

  const int tid  = threadIdx.x;
  const int lane = tid & 31;
  const int wave = tid >> 5;
  const int half = lane >> 4;
  const int m    = lane & 15;

  stage_weights(lwf1, Wn1, 8, 256, tid);
  stage_weights(lwf2, Wn2, 4, 128, tid);
  __syncthreads();

  const int tile = blockIdx.x * 8 + wave;
  const int base = tile * 16;
  const bool active = base < N;
  _Float16* sm = scratch + wave * (16 * DH);

  v8f acc[8];

  if (active) {
    int node = base + m;
    if (node >= N) node = N - 1;
    const float* rowX = x + (size_t)node * DH;
    const float* rowM = msum + (size_t)node * DH;
    const float inv = 1.0f / fmaxf(deg[node], 1.0f);

#pragma unroll
    for (int t = 0; t < 8; ++t) {
      float bias = bn1[t * 16 + m];
      v8f c;
#pragma unroll
      for (int r = 0; r < 8; ++r) c[r] = bias;
      acc[t] = c;
    }
#pragma unroll
    for (int kt = 0; kt < 8; ++kt) {
      v16h a;
      if (kt < 4) {
        a = load_a_row(rowX, kt * 32, half);
      } else {
        int cb = (kt - 4) * 32;
#pragma unroll
        for (int e = 0; e < 16; ++e) {
          int K = cb + ((e & 8) << 1) + half * 8 + (e & 7);
          a[e] = (_Float16)(rowM[K] * inv);
        }
      }
#pragma unroll
      for (int nt = 0; nt < 8; ++nt)
        acc[nt] = wmma_f16(a, load_b_frag(lwf1, kt, nt, lane), acc[nt]);
    }
#pragma unroll
    for (int t = 0; t < 8; ++t)
#pragma unroll
      for (int r = 0; r < 8; ++r)
        sm[(r + 8 * half) * DH + t * 16 + m] = (_Float16)silu_f(acc[t][r]);
  }
  __syncthreads();

  if (active) {
#pragma unroll
    for (int t = 0; t < 8; ++t) {
      float bias = bn2[t * 16 + m];
      v8f c;
#pragma unroll
      for (int r = 0; r < 8; ++r) c[r] = bias;
      acc[t] = c;
    }
#pragma unroll
    for (int kt = 0; kt < 4; ++kt) {
      v16h a2 = load_a_frag_sm(sm, kt, lane);
#pragma unroll
      for (int nt = 0; nt < 8; ++nt)
        acc[nt] = wmma_f16(a2, load_b_frag(lwf2, kt, nt, lane), acc[nt]);
    }
    // x_out stores (C layout -> row-major global)
#pragma unroll
    for (int t = 0; t < 8; ++t)
#pragma unroll
      for (int r = 0; r < 8; ++r) {
        int node = base + r + 8 * half;
        if (node < N) xout[(size_t)node * DH + t * 16 + m] = acc[t][r];
      }
    // pos_out = pos + coord_acc / deg
    if (half == 0) {
      int node = base + m;
      if (node < N) {
        float inv = 1.0f / fmaxf(deg[node], 1.0f);
#pragma unroll
        for (int c = 0; c < 3; ++c)
          posout[node * 3 + c] = pos[node * 3 + c] + cacc[node * 3 + c] * inv;
      }
    }
  }
}

// ---------------------------------------------------------------------------
extern "C" void kernel_launch(void* const* d_in, const int* in_sizes, int n_in,
                              void* d_out, int out_size, void* d_ws,
                              size_t ws_size, hipStream_t stream) {
  const float* x     = (const float*)d_in[0];
  const float* pos   = (const float*)d_in[1];
  const int*   ei    = (const int*)d_in[2];   // [2,E]: src row then dst row
  const float* eattr = (const float*)d_in[3];
  const float* s     = (const float*)d_in[4];
  const float* We1   = (const float*)d_in[5];
  const float* be1   = (const float*)d_in[6];
  const float* We2   = (const float*)d_in[7];
  const float* be2   = (const float*)d_in[8];
  const float* Wn1   = (const float*)d_in[9];
  const float* bn1   = (const float*)d_in[10];
  const float* Wn2   = (const float*)d_in[11];
  const float* bn2   = (const float*)d_in[12];
  const float* Wc    = (const float*)d_in[13];
  const float* bc    = (const float*)d_in[14];

  const int N = in_sizes[0] / 128;
  const int E = in_sizes[3];

  // workspace: m_sum[N,128] | deg[N] | coord_acc[N,3]
  float* msum = (float*)d_ws;
  float* deg  = msum + (size_t)N * 128;
  float* cacc = deg + N;

  float* xout   = (float*)d_out;
  float* posout = xout + (size_t)N * 128;

  long long zn = (long long)N * 132;  // 128 + 1 + 3 floats per node
  int zb = (int)((zn + 255) / 256);
  if (zb > 4096) zb = 4096;
  egnn_zero_kernel<<<zb, 256, 0, stream>>>(msum, zn);

  const int eblocks = (E + 127) / 128;  // 8 waves x 16 edges per block
  const size_t elds = (size_t)(9 * 8 * 512 + 4 * 8 * 512 + 8 * 2048) *
                      sizeof(_Float16);  // 139264 B
  egnn_edge_kernel<<<eblocks, 256, elds, stream>>>(
      x, pos, ei, eattr, s, We1, be1, We2, be2, Wc, bc, msum, deg, cacc, E);

  const int nblocks = (N + 127) / 128;
  const size_t nlds = (size_t)(8 * 8 * 512 + 4 * 8 * 512 + 8 * 2048) *
                      sizeof(_Float16);  // 131072 B
  egnn_node_kernel<<<nblocks, 256, nlds, stream>>>(
      x, pos, Wn1, bn1, Wn2, bn2, msum, deg, cacc, xout, posout, N);
}
